// CustomLSTM_78434692759652
// MI455X (gfx1250) — matrix operand order repaired and checked by reading, not compiled, not run on hardware
//
#include <hip/hip_runtime.h>
#include <hip/hip_fp16.h>

#define I_SZ   512
#define H_SZ   1024
#define G_SZ   4096      // 4*H
#define K_TOT  1536      // I + H
#define B_SZ   32
#define S_SZ   512
#define NWG    64        // persistent workgroups; each owns 16 hidden columns
#define WSTR   1544      // padded LDS column stride in f16 (1536 + 8) -> conflict-free

typedef __attribute__((ext_vector_type(16))) _Float16 v16h;
typedef __attribute__((ext_vector_type(8)))  _Float16 v8h;
typedef __attribute__((ext_vector_type(8)))  float    v8f;

// ---------------------------------------------------------------------------
// Prep 1: pack [W;U] for all 4 gates into column-major f16  WT[n][k],
//         n = gate*1024 + hidden_col (4096 cols), k = 0..1535 (I then H).
// ---------------------------------------------------------------------------
__global__ void pack_weights(const float* __restrict__ w_i, const float* __restrict__ u_i,
                             const float* __restrict__ w_f, const float* __restrict__ u_f,
                             const float* __restrict__ w_c, const float* __restrict__ u_c,
                             const float* __restrict__ w_o, const float* __restrict__ u_o,
                             _Float16* __restrict__ WT) {
  int idx = blockIdx.x * 256 + threadIdx.x;       // n*K_TOT + k
  if (idx >= G_SZ * K_TOT) return;
  int k = idx % K_TOT;
  int n = idx / K_TOT;
  int g = n >> 10;                                 // gate: 0=i,1=f,2=c,3=o
  int j = n & 1023;                                // hidden column
  const float* w = (g == 0) ? w_i : (g == 1) ? w_f : (g == 2) ? w_c : w_o;
  const float* u = (g == 0) ? u_i : (g == 1) ? u_f : (g == 2) ? u_c : u_o;
  float v = (k < I_SZ) ? w[(size_t)k * H_SZ + j] : u[(size_t)(k - I_SZ) * H_SZ + j];
  WT[idx] = (_Float16)v;
}

// ---------------------------------------------------------------------------
// Prep 2: transpose X [B][S][I] f32  ->  Xh [S][B][I] f16
// ---------------------------------------------------------------------------
__global__ void pack_x(const float* __restrict__ X, _Float16* __restrict__ Xh) {
  int idx = blockIdx.x * 256 + threadIdx.x;       // t*B*I + b*I + i
  if (idx >= S_SZ * B_SZ * I_SZ) return;
  int i = idx & (I_SZ - 1);
  int b = (idx >> 9) & 31;
  int t = idx >> 14;
  Xh[idx] = (_Float16)X[((size_t)b * S_SZ + t) * I_SZ + i];
}

__device__ __forceinline__ float sigmoidf(float x) { return 1.f / (1.f + __expf(-x)); }

// ---------------------------------------------------------------------------
// Persistent recurrence kernel. 64 WGs x 256 threads (8 wave32).
// Weight slice (64 gate-cols x 1536 K, f16) staged ONCE into LDS via
// global_load_async_to_lds_b128, reused for all 512 timesteps.
// Wave w: gate = w&3, mrow = w>>2; one 16x16 C tile per wave,
// 48 x v_wmma_f32_16x16x32_f16 per step (A from L2, B from LDS).
// Grid-wide sync per timestep via atomic generation counters.
// ---------------------------------------------------------------------------
__global__ void __launch_bounds__(256, 1)
lstm_persistent(const _Float16* __restrict__ WT,
                const _Float16* __restrict__ Xh,
                _Float16* __restrict__ hbuf,          // 2 parity buffers [B][H] f16
                unsigned int* __restrict__ syncv,     // [S] counters, pre-zeroed
                const float* __restrict__ b_i, const float* __restrict__ b_f,
                const float* __restrict__ b_c, const float* __restrict__ b_o,
                float* __restrict__ out)              // hidden_seq[B][S][H] + h[B][H] + c[B][H]
{
  __shared__ __align__(16) _Float16 wlds[64 * WSTR];  // 197,632 B weight slice
  __shared__ float cex[8][16][16];                    // C-tile exchange (gate combine)
  __shared__ float cst[512];                          // cell state: 32 batch x 16 cols

  const int tid  = threadIdx.x;
  const int wave = tid >> 5;
  const int lane = tid & 31;
  const int gate = wave & 3;
  const int mrow = wave >> 2;
  const int wgj  = blockIdx.x * 16;                   // hidden column base of this WG

  cst[tid] = 0.f; cst[tid + 256] = 0.f;               // c0 = 0

  // ---- one-time async stage: WG's 64 weight columns -> LDS (ASYNCcnt path) ----
  {
    const unsigned lbase = (unsigned)(size_t)(void*)wlds;   // LDS byte address of wlds
    for (int c = tid; c < 64 * 192; c += 256) {             // 16-byte chunks
      int cl = c / 192;                                     // LDS column: g*16 + l
      int kc = c % 192;                                     // chunk along K
      int g  = cl >> 4, l = cl & 15;
      const _Float16* gsrc = WT + ((size_t)g * H_SZ + wgj + l) * K_TOT + (size_t)kc * 8;
      unsigned loff = lbase + (unsigned)cl * (WSTR * 2) + (unsigned)kc * 16;
      asm volatile("global_load_async_to_lds_b128 %0, %1, off"
                   :: "v"(loff), "v"(gsrc) : "memory");
    }
    asm volatile("s_wait_asynccnt 0" ::: "memory");
  }

  // Elementwise assignment: each thread owns 2 (batch, col) pairs.
  const int p0 = tid * 2, p1 = p0 + 1;
  const int j0 = wgj + (p0 & 15), j1 = wgj + (p1 & 15);
  const int eb0 = p0 >> 4,        eb1 = p1 >> 4;      // batch index 0..31
  const float bi0 = b_i[j0], bf0 = b_f[j0], bc0 = b_c[j0], bo0 = b_o[j0];
  const float bi1 = b_i[j1], bf1 = b_f[j1], bc1 = b_c[j1], bo1 = b_o[j1];

  // WMMA operand addressing (ISA 16-bit A/B VGPR layouts, wave32)
  const int am   = mrow * 16 + (lane & 15);           // A row (batch)
  const int akof = (lane >> 4) * 8;                   // A: lane<16 -> K 0..7 & 16..23
  const int bkof = (lane >> 4) * 8;                   // B: lane<16 -> K 0..7 & 16..23
  const _Float16* blds = wlds + (size_t)(gate * 16 + (lane & 15)) * WSTR + bkof;

  __syncthreads();                                    // weights visible to all waves

  for (int t = 0; t < S_SZ; ++t) {
    const _Float16* hread  = hbuf + (size_t)(t & 1) * (B_SZ * H_SZ);
    _Float16*       hwrite = hbuf + (size_t)((t + 1) & 1) * (B_SZ * H_SZ);

    v8f acc = {};
    // ---- input projection slice: K = 0..511 (A from Xh[t], B from LDS) ----
    {
      const _Float16* arow = Xh + ((size_t)t * B_SZ + am) * I_SZ;
      #pragma unroll 4
      for (int k0 = 0; k0 < I_SZ; k0 += 32) {
        v16h a, b;
        *(v8h*)&a       = *(const v8h*)(arow + k0 + akof);
        *((v8h*)&a + 1) = *(const v8h*)(arow + k0 + akof + 16);
        *(v8h*)&b       = *(const v8h*)(blds + k0);
        *((v8h*)&b + 1) = *(const v8h*)(blds + k0 + 16);
        acc = __builtin_amdgcn_wmma_f32_16x16x32_f16(false, a, false, b,
                                                     (short)0, acc, false, false);
      }
    }
    // ---- recurrent slice: K = 512..1535 (A from h_{t-1}, B from LDS) ----
    {
      const _Float16* arow = hread + (size_t)am * H_SZ;
      const _Float16* bh   = blds + I_SZ;
      #pragma unroll 4
      for (int k0 = 0; k0 < H_SZ; k0 += 32) {
        v16h a, b;
        *(v8h*)&a       = *(const v8h*)(arow + k0 + akof);
        *((v8h*)&a + 1) = *(const v8h*)(arow + k0 + akof + 16);
        *(v8h*)&b       = *(const v8h*)(bh + k0);
        *((v8h*)&b + 1) = *(const v8h*)(bh + k0 + 16);
        acc = __builtin_amdgcn_wmma_f32_16x16x32_f16(false, a, false, b,
                                                     (short)0, acc, false, false);
      }
    }

    // ---- share C tiles across waves (gate combine) ----
    {
      const int mb = (lane >> 4) * 8;                 // C layout: lanes>=16 hold M=8..15
      const int nn = lane & 15;
      #pragma unroll
      for (int r = 0; r < 8; ++r) cex[wave][mb + r][nn] = acc[r];
    }
    __syncthreads();

    // ---- LSTM cell elementwise: 2 (batch,col) pairs per thread ----
    {
      int mr = eb0 >> 4, ml = eb0 & 15, nn = p0 & 15;
      float gi = cex[mr*4 + 0][ml][nn] + bi0;
      float gf = cex[mr*4 + 1][ml][nn] + bf0;
      float gc = cex[mr*4 + 2][ml][nn] + bc0;
      float go = cex[mr*4 + 3][ml][nn] + bo0;
      float cn = sigmoidf(gf) * cst[p0] + sigmoidf(gi) * tanhf(gc);
      float hn = sigmoidf(go) * tanhf(cn);
      cst[p0] = cn;
      out[(size_t)eb0 * (S_SZ * H_SZ) + (size_t)t * H_SZ + j0] = hn;
      hwrite[eb0 * H_SZ + j0] = (_Float16)hn;
      if (t == S_SZ - 1) {
        out[(size_t)B_SZ * S_SZ * H_SZ + (size_t)eb0 * H_SZ + j0] = hn;
        out[(size_t)B_SZ * S_SZ * H_SZ + (size_t)B_SZ * H_SZ + (size_t)eb0 * H_SZ + j0] = cn;
      }
    }
    {
      int mr = eb1 >> 4, ml = eb1 & 15, nn = p1 & 15;
      float gi = cex[mr*4 + 0][ml][nn] + bi1;
      float gf = cex[mr*4 + 1][ml][nn] + bf1;
      float gc = cex[mr*4 + 2][ml][nn] + bc1;
      float go = cex[mr*4 + 3][ml][nn] + bo1;
      float cn = sigmoidf(gf) * cst[p1] + sigmoidf(gi) * tanhf(gc);
      float hn = sigmoidf(go) * tanhf(cn);
      cst[p1] = cn;
      out[(size_t)eb1 * (S_SZ * H_SZ) + (size_t)t * H_SZ + j1] = hn;
      hwrite[eb1 * H_SZ + j1] = (_Float16)hn;
      if (t == S_SZ - 1) {
        out[(size_t)B_SZ * S_SZ * H_SZ + (size_t)eb1 * H_SZ + j1] = hn;
        out[(size_t)B_SZ * S_SZ * H_SZ + (size_t)B_SZ * H_SZ + (size_t)eb1 * H_SZ + j1] = cn;
      }
    }

    // ---- grid-wide generation barrier for step t ----
    __threadfence();
    if (tid == 0) {
      atomicAdd(&syncv[t], 1u);
      while (__hip_atomic_load(&syncv[t], __ATOMIC_ACQUIRE,
                               __HIP_MEMORY_SCOPE_AGENT) < (unsigned)NWG) {
        __builtin_amdgcn_s_sleep(2);
      }
    }
    __syncthreads();   // also separates cex reads from next step's writes
  }
}

// ---------------------------------------------------------------------------
extern "C" void kernel_launch(void* const* d_in, const int* in_sizes, int n_in,
                              void* d_out, int out_size, void* d_ws, size_t ws_size,
                              hipStream_t stream) {
  (void)in_sizes; (void)n_in; (void)out_size; (void)ws_size;
  const float* X   = (const float*)d_in[0];
  const float* w_i = (const float*)d_in[1];
  const float* u_i = (const float*)d_in[2];
  const float* b_i = (const float*)d_in[3];
  const float* w_f = (const float*)d_in[4];
  const float* u_f = (const float*)d_in[5];
  const float* b_f = (const float*)d_in[6];
  const float* w_c = (const float*)d_in[7];
  const float* u_c = (const float*)d_in[8];
  const float* b_c = (const float*)d_in[9];
  const float* w_o = (const float*)d_in[10];
  const float* u_o = (const float*)d_in[11];
  const float* b_o = (const float*)d_in[12];

  char* ws = (char*)d_ws;
  const size_t WT_BYTES = (size_t)G_SZ * K_TOT * sizeof(_Float16);       // 12,582,912
  const size_t XH_BYTES = (size_t)S_SZ * B_SZ * I_SZ * sizeof(_Float16); // 16,777,216
  const size_t HB_BYTES = (size_t)2 * B_SZ * H_SZ * sizeof(_Float16);    // 131,072
  _Float16* WT   = (_Float16*)ws;
  _Float16* Xh   = (_Float16*)(ws + WT_BYTES);
  _Float16* hbuf = (_Float16*)(ws + WT_BYTES + XH_BYTES);
  unsigned int* syncv = (unsigned int*)(ws + WT_BYTES + XH_BYTES + HB_BYTES);

  // Re-initialize mutable state every launch (graph-replay deterministic).
  hipMemsetAsync(hbuf, 0, HB_BYTES, stream);                 // h0 = 0 (both parities)
  hipMemsetAsync(syncv, 0, (size_t)S_SZ * sizeof(unsigned int), stream);

  pack_weights<<<(G_SZ * K_TOT + 255) / 256, 256, 0, stream>>>(
      w_i, u_i, w_f, u_f, w_c, u_c, w_o, u_o, WT);
  pack_x<<<(S_SZ * B_SZ * I_SZ + 255) / 256, 256, 0, stream>>>(X, Xh);
  lstm_persistent<<<NWG, 256, 0, stream>>>(WT, Xh, hbuf, syncv,
                                           b_i, b_f, b_c, b_o, (float*)d_out);
}
